// CustomMultiheadAttention_8203387535729
// MI455X (gfx1250) — compile-verified, matrix operand order checked by reading, not run on hardware
//
#include <hip/hip_runtime.h>
#include <hip/hip_bf16.h>
#include <cmath>

// ---------------------------------------------------------------------------
// CustomMultiheadAttention for MI455X (gfx1250, wave32, WMMA f16->f32)
//   B=2, S=2048, D=1024, H=16, hd=64
//   kernel 0: one-shot fp32 -> f16 conversion (weights + activations)
//   kernel 1: qkv projection  (all-f16 GEMM -> f16 headmajor Q/K/V in ws)
//   kernel 2: fused causal flash attention (async-to-LDS K staging)
//   kernel 3: output projection (f16 ctx -> fp32 out)
// ---------------------------------------------------------------------------

#define BATCH   2
#define SEQ     2048
#define DMODEL  1024
#define NHEADS  16
#define HDIM    64
#define ROWS    (BATCH * SEQ)            // 4096 flattened rows

typedef __attribute__((ext_vector_type(16))) _Float16 v16h;
typedef __attribute__((ext_vector_type(4)))  _Float16 v4h;
typedef __attribute__((ext_vector_type(8)))  float    v8f;

// D = A*B + C, 16x16x32 f16 -> f32
static __device__ __forceinline__ v8f wmma_f16(v16h a, v16h b, v8f c) {
    return __builtin_amdgcn_wmma_f32_16x16x32_f16(
        false, a, false, b, (short)0, c, false, false);
}

// ---------------------------------------------------------------------------
// Kernel 0: fp32 -> f16 bulk convert (4 floats / thread)
// ---------------------------------------------------------------------------
__global__ void __launch_bounds__(256)
mha_cvt_f16_kernel(const float* __restrict__ src, _Float16* __restrict__ dst)
{
    const int idx = (blockIdx.x * 256 + threadIdx.x) * 4;
    const float4 v = *(const float4*)(src + idx);
    v4h h;
    h[0] = (_Float16)v.x;  h[1] = (_Float16)v.y;
    h[2] = (_Float16)v.z;  h[3] = (_Float16)v.w;
    *(v4h*)(dst + idx) = h;
}

// ---------------------------------------------------------------------------
// Kernel 1: Y = X @ W^T + bias, all-f16 operands, f16 head-major out [B,H,S,hd]
//   grid = (ROWS/16/8, DMODEL/64) ; block = 256 (8 waves)
// ---------------------------------------------------------------------------
__global__ void __launch_bounds__(256)
mha_qkv_proj_kernel(const _Float16* __restrict__ X, const _Float16* __restrict__ W,
                    const float* __restrict__ bias, _Float16* __restrict__ Y)
{
    const int lane  = threadIdx.x & 31;
    const int wave  = threadIdx.x >> 5;
    const int ln    = lane & 15;          // N column / M row within tile
    const int hi    = lane >> 4;          // half-wave select
    const int baseK = hi * 8;             // ISA A/B fragment K offset

    const int mtile  = blockIdx.x * 8 + wave;     // 0..255
    const int row    = mtile * 16 + ln;           // global A row (lane's M)
    const int ngroup = blockIdx.y;                // head index (64 cols)

    v8f c[4] = {};
    const _Float16* __restrict__ pa_base = X + (size_t)row * DMODEL;

    for (int kk = 0; kk < DMODEL; kk += 32) {
        // ---- A fragment: X[row][kk + baseK + {0..7, 16..23}] ----
        const _Float16* pa = pa_base + kk + baseK;
        __builtin_prefetch(pa + 64, 0, 0);        // global_prefetch_b8
        v16h a;
        #pragma unroll
        for (int i = 0; i < 8; ++i) {
            a[i]     = pa[i];
            a[8 + i] = pa[16 + i];
        }
        // ---- 4 B fragments (B = W^T, so b[i] = W[ncol][k]) ----
        #pragma unroll
        for (int t = 0; t < 4; ++t) {
            const int ncol = ngroup * 64 + t * 16 + ln;
            const _Float16* pb = W + (size_t)ncol * DMODEL + kk + baseK;
            v16h b;
            #pragma unroll
            for (int i = 0; i < 8; ++i) {
                b[i]     = pb[i];
                b[8 + i] = pb[16 + i];
            }
            c[t] = wmma_f16(a, b, c[t]);
        }
    }

    // ---- epilogue: bias + store f16 head-major [B,H,S,hd] ----
    #pragma unroll
    for (int t = 0; t < 4; ++t) {
        const int ncol = ngroup * 64 + t * 16 + ln;
        const float bb = bias[ncol];
        #pragma unroll
        for (int r = 0; r < 8; ++r) {
            const int   grow = mtile * 16 + r + hi * 8;   // flattened row
            const int   bidx = grow >> 11;                // /2048
            const int   s    = grow & 2047;
            const int   d    = t * 16 + ln;
            const size_t dst = (((size_t)bidx * NHEADS + ngroup) * SEQ + s) * HDIM + d;
            Y[dst] = (_Float16)(c[t][r] + bb);
        }
    }
}

// ---------------------------------------------------------------------------
// Kernel 2: fused causal flash attention over f16 head-major Q/K/V
//   grid = (SEQ/128, B*H) ; block = 256 (8 waves, each owns 16 q rows)
//   K tile staged memory->LDS with global_load_async_to_lds_b128 (ASYNCcnt)
// ---------------------------------------------------------------------------
__global__ void __launch_bounds__(256)
mha_flash_attn_kernel(const _Float16* __restrict__ Qh,
                      const _Float16* __restrict__ Kh,
                      const _Float16* __restrict__ Vh,
                      _Float16* __restrict__ Ctx)
{
    const int lane  = threadIdx.x & 31;
    const int wave  = threadIdx.x >> 5;
    const int ln    = lane & 15;
    const int hi    = lane >> 4;
    const int baseK = hi * 8;

    const int bh   = blockIdx.y;          // 0..31
    const int bidx = bh >> 4;
    const int head = bh & 15;
    const size_t hoff = (size_t)bh * SEQ * HDIM;
    const _Float16* __restrict__ Q = Qh + hoff;
    const _Float16* __restrict__ K = Kh + hoff;
    const _Float16* __restrict__ V = Vh + hoff;

    const int qbase = blockIdx.x * 128;       // block's first q row
    const int qrowA = qbase + wave * 16 + ln; // this lane's A-fragment row

    // LDS staging (per block): K tile, transposed V tile, per-wave P tiles
    __shared__ __align__(16) _Float16 Ksh[32][72];   // [kv][d]   (+8 pad)
    __shared__ __align__(16) _Float16 Vts[64][40];   // [d][kv]   transposed
    __shared__ __align__(16) _Float16 Psh[8][16][40];// per-wave P staging

    // per-thread staging coordinates (one 16B chunk each; 256 thr = 4KB tile)
    const int krow = threadIdx.x >> 3;        // 0..31
    const int kcol = (threadIdx.x & 7) * 8;   // 0..56
    // LDS byte address of this thread's K destination (low 32 bits of generic)
    const unsigned ldsK = (unsigned)(uintptr_t)&Ksh[krow][kcol];

    // Q fragments for hd=64: two K-chunks of 32
    v16h qa[2];
    {
        const _Float16* pq = Q + (size_t)qrowA * HDIM;
        #pragma unroll
        for (int ck = 0; ck < 2; ++ck)
            #pragma unroll
            for (int i = 0; i < 8; ++i) {
                qa[ck][i]     = pq[ck * 32 + baseK + i];
                qa[ck][8 + i] = pq[ck * 32 + baseK + 16 + i];
            }
    }

    v8f  o[4] = {};
    float mrow[8], lrow[8];
    #pragma unroll
    for (int r = 0; r < 8; ++r) { mrow[r] = -1e30f; lrow[r] = 0.0f; }

    const float scale  = 0.125f;             // 1/sqrt(64)
    const int   kv_end = qbase + 128;        // causal limit for whole block

    for (int kv0 = 0; kv0 < kv_end; kv0 += 32) {
        // ---- K tile: async copy global -> LDS (no VGPR round-trip) ----
        {
            const _Float16* gk = K + (size_t)(kv0 + krow) * HDIM + kcol;
            asm volatile("global_load_async_to_lds_b128 %0, %1, off"
                         :: "v"(ldsK), "v"(gk) : "memory");
        }
        // ---- V tile: load + transpose into LDS (manual path) ----
        {
            const _Float16* gv = V + (size_t)(kv0 + krow) * HDIM + kcol;
            __builtin_prefetch(gv + 32 * HDIM, 0, 0);
            #pragma unroll
            for (int i = 0; i < 8; ++i)
                Vts[kcol + i][krow] = gv[i];
        }
        asm volatile("s_wait_asynccnt 0" ::: "memory");
        __syncthreads();

        // ---- scores: S(16x32) = Q(16x64) . K^T, two 16x16 D-fragments ----
        v8f s0 = {}, s1 = {};
        #pragma unroll
        for (int ck = 0; ck < 2; ++ck) {
            v16h b0, b1;
            #pragma unroll
            for (int i = 0; i < 8; ++i) {
                b0[i]     = Ksh[ln     ][ck * 32 + baseK + i];
                b0[8 + i] = Ksh[ln     ][ck * 32 + baseK + 16 + i];
                b1[i]     = Ksh[ln + 16][ck * 32 + baseK + i];
                b1[8 + i] = Ksh[ln + 16][ck * 32 + baseK + 16 + i];
            }
            s0 = wmma_f16(qa[ck], b0, s0);
            s1 = wmma_f16(qa[ck], b1, s1);
        }

        // ---- online softmax (rows live across 16-lane groups) ----
        #pragma unroll
        for (int r = 0; r < 8; ++r) {
            const int rrow = qbase + wave * 16 + r + hi * 8;
            const int c0   = kv0 + ln;
            const int c1   = c0 + 16;
            float v0 = (c0 <= rrow) ? s0[r] * scale : -1e30f;
            float v1 = (c1 <= rrow) ? s1[r] * scale : -1e30f;

            float mx = fmaxf(v0, v1);
            #pragma unroll
            for (int msk = 1; msk < 16; msk <<= 1)
                mx = fmaxf(mx, __shfl_xor(mx, msk, 32));

            const float mnew  = fmaxf(mrow[r], mx);
            const float alpha = __expf(mrow[r] - mnew);
            const float p0    = __expf(v0 - mnew);
            const float p1    = __expf(v1 - mnew);

            float rs = p0 + p1;
            #pragma unroll
            for (int msk = 1; msk < 16; msk <<= 1)
                rs += __shfl_xor(rs, msk, 32);

            lrow[r] = lrow[r] * alpha + rs;
            mrow[r] = mnew;
            #pragma unroll
            for (int t = 0; t < 4; ++t) o[t][r] *= alpha;

            // stash P in D-layout; re-read below in A-layout
            Psh[wave][r + hi * 8][ln]      = (_Float16)p0;
            Psh[wave][r + hi * 8][ln + 16] = (_Float16)p1;
        }
        __builtin_amdgcn_wave_barrier();   // keep LDS write->read ordered

        // ---- O(16x64) += P(16x32) . V(32x64) ----
        v16h ap;
        #pragma unroll
        for (int i = 0; i < 8; ++i) {
            ap[i]     = Psh[wave][ln][baseK + i];
            ap[8 + i] = Psh[wave][ln][baseK + 16 + i];
        }
        #pragma unroll
        for (int t = 0; t < 4; ++t) {
            v16h bv;
            #pragma unroll
            for (int i = 0; i < 8; ++i) {
                bv[i]     = Vts[t * 16 + ln][baseK + i];
                bv[8 + i] = Vts[t * 16 + ln][baseK + 16 + i];
            }
            o[t] = wmma_f16(ap, bv, o[t]);
        }
        __syncthreads();   // protect Ksh/Vts before next staging
    }

    // ---- normalize and write ctx as f16 [B,S,D] ----
    #pragma unroll
    for (int r = 0; r < 8; ++r) {
        const float inv  = 1.0f / lrow[r];
        const int   rrow = qbase + wave * 16 + r + hi * 8;
        const size_t rowoff = ((size_t)bidx * SEQ + rrow) * DMODEL + head * HDIM;
        #pragma unroll
        for (int t = 0; t < 4; ++t)
            Ctx[rowoff + t * 16 + ln] = (_Float16)(o[t][r] * inv);
    }
}

// ---------------------------------------------------------------------------
// Kernel 3: out = ctx @ o_w^T + o_b   (f16 operands -> fp32 out)
// ---------------------------------------------------------------------------
__global__ void __launch_bounds__(256)
mha_out_proj_kernel(const _Float16* __restrict__ Xh, const _Float16* __restrict__ W,
                    const float* __restrict__ bias, float* __restrict__ Out)
{
    const int lane  = threadIdx.x & 31;
    const int wave  = threadIdx.x >> 5;
    const int ln    = lane & 15;
    const int hi    = lane >> 4;
    const int baseK = hi * 8;

    const int mtile  = blockIdx.x * 8 + wave;
    const int row    = mtile * 16 + ln;
    const int ngroup = blockIdx.y;

    v8f c[4] = {};
    const _Float16* __restrict__ pa_base = Xh + (size_t)row * DMODEL;

    for (int kk = 0; kk < DMODEL; kk += 32) {
        const _Float16* pa = pa_base + kk + baseK;
        __builtin_prefetch(pa + 64, 0, 0);
        v16h a;
        #pragma unroll
        for (int i = 0; i < 8; ++i) {
            a[i]     = pa[i];
            a[8 + i] = pa[16 + i];
        }
        #pragma unroll
        for (int t = 0; t < 4; ++t) {
            const int ncol = ngroup * 64 + t * 16 + ln;
            const _Float16* pb = W + (size_t)ncol * DMODEL + kk + baseK;
            v16h b;
            #pragma unroll
            for (int i = 0; i < 8; ++i) {
                b[i]     = pb[i];
                b[8 + i] = pb[16 + i];
            }
            c[t] = wmma_f16(a, b, c[t]);
        }
    }

    #pragma unroll
    for (int t = 0; t < 4; ++t) {
        const int ncol = ngroup * 64 + t * 16 + ln;
        const float bb = bias[ncol];
        #pragma unroll
        for (int r = 0; r < 8; ++r) {
            const int grow = mtile * 16 + r + hi * 8;
            Out[(size_t)grow * DMODEL + ncol] = c[t][r] + bb;
        }
    }
}

// ---------------------------------------------------------------------------
extern "C" void kernel_launch(void* const* d_in, const int* in_sizes, int n_in,
                              void* d_out, int out_size, void* d_ws, size_t ws_size,
                              hipStream_t stream)
{
    const float* query = (const float*)d_in[0];
    const float* key   = (const float*)d_in[1];
    const float* value = (const float*)d_in[2];
    // d_in[3] = attn_mask (causal; implicit in kernel 2)
    const float* q_w = (const float*)d_in[4];
    const float* q_b = (const float*)d_in[5];
    const float* k_w = (const float*)d_in[6];
    const float* k_b = (const float*)d_in[7];
    const float* v_w = (const float*)d_in[8];
    const float* v_b = (const float*)d_in[9];
    const float* o_w = (const float*)d_in[10];
    const float* o_b = (const float*)d_in[11];
    float* out = (float*)d_out;

    // workspace layout (f16 elements):
    //   Qh/Kh/Vh headmajor (4M each) | Ctx (4M) | Xq/Xk/Xv (4M each) | 4 weights (1M each)
    const size_t asz = (size_t)ROWS * DMODEL;          // 4M activation elems
    const size_t wsz = (size_t)DMODEL * DMODEL;        // 1M weight elems
    _Float16* Qh  = (_Float16*)d_ws;
    _Float16* Kh  = Qh + asz;
    _Float16* Vh  = Kh + asz;
    _Float16* Ctx = Vh + asz;
    _Float16* Xq  = Ctx + asz;
    _Float16* Xk  = Xq + asz;
    _Float16* Xv  = Xk + asz;
    _Float16* Wq  = Xv + asz;
    _Float16* Wk  = Wq + wsz;
    _Float16* Wv  = Wk + wsz;
    _Float16* Wo  = Wv + wsz;

    const dim3 blk(256);
    const dim3 gCvtA((unsigned)(asz / 1024));       // activations: 4096 blocks
    const dim3 gCvtW((unsigned)(wsz / 1024));       // weights:     1024 blocks
    const dim3 gProj(ROWS / 16 / 8, DMODEL / 64);   // (32, 16)
    const dim3 gAttn(SEQ / 128, BATCH * NHEADS);    // (16, 32)

    // one-shot fp32 -> f16 conversion
    mha_cvt_f16_kernel<<<gCvtA, blk, 0, stream>>>(query, Xq);
    mha_cvt_f16_kernel<<<gCvtA, blk, 0, stream>>>(key,   Xk);
    mha_cvt_f16_kernel<<<gCvtA, blk, 0, stream>>>(value, Xv);
    mha_cvt_f16_kernel<<<gCvtW, blk, 0, stream>>>(q_w, Wq);
    mha_cvt_f16_kernel<<<gCvtW, blk, 0, stream>>>(k_w, Wk);
    mha_cvt_f16_kernel<<<gCvtW, blk, 0, stream>>>(v_w, Wv);
    mha_cvt_f16_kernel<<<gCvtW, blk, 0, stream>>>(o_w, Wo);

    // projections -> fused attention -> output projection
    mha_qkv_proj_kernel<<<gProj, blk, 0, stream>>>(Xq, Wq, q_b, Qh);
    mha_qkv_proj_kernel<<<gProj, blk, 0, stream>>>(Xk, Wk, k_b, Kh);
    mha_qkv_proj_kernel<<<gProj, blk, 0, stream>>>(Xv, Wv, v_b, Vh);
    mha_flash_attn_kernel<<<gAttn, blk, 0, stream>>>(Qh, Kh, Vh, Ctx);
    mha_out_proj_kernel<<<gProj, blk, 0, stream>>>(Ctx, Wo, o_b, out);
}
